// LocalAttention1d_57621281243440
// MI455X (gfx1250) — matrix operand order repaired and checked by reading, not compiled, not run on hardware
//
#include <hip/hip_runtime.h>
#include <hip/hip_bf16.h>

// ---------------------------------------------------------------------------
// LocalAttention1d for MI455X (gfx1250, wave32, WMMA + TDM).
// Sizes fixed by the reference: B=16, T=1024, S=4096, q=c=p=512, D=3, W=7.
// ---------------------------------------------------------------------------

#define B_SZ   16
#define T_SZ   1024
#define S_SZ   4096
#define C_SZ   512             // q_size == c_size == p_size == 512
#define M_SZ   (B_SZ * T_SZ)   // 16384 rows
#define WIN    7

typedef __attribute__((ext_vector_type(16))) __bf16 v16bf;
typedef __attribute__((ext_vector_type(8)))  float  v8f;
typedef __attribute__((ext_vector_type(4)))  unsigned v4u_t;
typedef __attribute__((ext_vector_type(8)))  int      v8i_t;
typedef __attribute__((ext_vector_type(4)))  int      v4i_t;

union BFrag { v16bf v; unsigned u[8]; };

// round-to-nearest-even f32 -> bf16
static __device__ __forceinline__ unsigned short f2bf(float f) {
  union { float f; unsigned u; } t; t.f = f;
  unsigned u = t.u + 0x7FFFu + ((t.u >> 16) & 1u);
  return (unsigned short)(u >> 16);
}
static __device__ __forceinline__ unsigned pack2(float lo, float hi) {
  return (unsigned)f2bf(lo) | ((unsigned)f2bf(hi) << 16);
}

// ---------------------------------------------------------------------------
// Kernel 1: convert W_p -> bf16 (row-major, rows contiguous in k=c) and
//           W_a -> bf16 transposed (wa_t[q][c] = W_a[c][q]) so the WMMA
//           B-fragment pair (k, k+1) is a single dword load for both GEMMs.
// ---------------------------------------------------------------------------
__global__ __launch_bounds__(256) void prep_kernel(
    const float* __restrict__ Wp, const float* __restrict__ Wa,
    unsigned short* __restrict__ wp_bf, unsigned short* __restrict__ wa_t) {
  int i = blockIdx.x * 256 + threadIdx.x;       // 0 .. 512*512-1
  int r = i >> 9;
  int c = i & 511;
  wp_bf[i] = f2bf(Wp[i]);                       // wp_bf[n][k] = W_p[n][k]
  wa_t[i]  = f2bf(Wa[c * C_SZ + r]);            // wa_t[q][c]  = W_a[c][q]
}

// ---------------------------------------------------------------------------
// Kernel 2: fused dual GEMM over a shared A tile staged by the TDM.
//   accP = c_t @ W_p^T   (-> tanh, dot V_p, sigmoid -> p_t)
//   accA = c_t @ W_a     (-> u, stored f32)
// One block = 16 rows of c_t; 8 waves each own a 64-wide N slab.
// ---------------------------------------------------------------------------
__global__ __launch_bounds__(256) void fused_gemm_kernel(
    const float* __restrict__ c_t, const float* __restrict__ V_p,
    const unsigned short* __restrict__ wp_bf,
    const unsigned short* __restrict__ wa_t,
    float* __restrict__ u, float* __restrict__ p_t) {
  __shared__ float    atile_f32[16 * 512];  // 32 KB raw tile (TDM destination)
  __shared__ unsigned atile[16 * 260];      // packed bf16 pairs, padded stride
  __shared__ float    red[16];

  const int tid = threadIdx.x;
  const int m0  = blockIdx.x * 16;
  if (tid < 16) red[tid] = 0.0f;

  // ---- Stage 16x512 f32 tile of c_t into LDS via the Tensor Data Mover ---
#if __has_builtin(__builtin_amdgcn_tensor_load_to_lds)
  if (tid == 0) {
    // Tensor DMA descriptor (D#), ISA 8.3/8.4: 2D tensor 16384x512 f32,
    // tile = 16 rows x 512 cols starting at row m0.
    const unsigned long long ga =
        (unsigned long long)(c_t + (unsigned long long)m0 * C_SZ);
    const unsigned lds_addr = (unsigned)(unsigned long long)(void*)atile_f32;

    v4u_t g0;
    g0[0] = 1u;                                   // count=1, user mode
    g0[1] = lds_addr;                             // lds_addr [63:32]
    g0[2] = (unsigned)(ga & 0xFFFFFFFFu);         // global_addr [95:64]
    g0[3] = (unsigned)((ga >> 32) & 0x01FFFFFFu)  // global_addr [120:96]
          | (2u << 30);                           // type=2 ("image") [127:126]

    v8i_t g1;
    g1[0] = (int)(2u << 16);                      // data_size=2 (4 bytes)
    g1[1] = (int)((unsigned)C_SZ << 16);          // tensor_dim0[15:0] @ [63:48]
    g1[2] = (int)(((unsigned)C_SZ >> 16)          // tensor_dim0[31:16]
          | ((unsigned)M_SZ << 16));              // tensor_dim1[15:0]
    g1[3] = (int)(((unsigned)M_SZ >> 16)          // tensor_dim1[31:16]
          | ((unsigned)C_SZ << 16));              // tile_dim0=512 @ [127:112]
    g1[4] = 16;                                   // tile_dim1=16, tile_dim2=0
    g1[5] = C_SZ;                                 // tensor_dim0_stride low32
    g1[6] = (int)((unsigned)C_SZ << 16);          // dim1_stride[15:0] @ [223:208]
    g1[7] = 0;                                    // dim1_stride high bits
    v4i_t g2 = {0, 0, 0, 0};
    v4i_t g3 = {0, 0, 0, 0};
    v8i_t g4 = {0, 0, 0, 0, 0, 0, 0, 0};          // unused 5th group (zeros)
    // amdgpu-toolchain (clang-23) 6-arg form.
    __builtin_amdgcn_tensor_load_to_lds(g0, g1, g2, g3, g4, 0);
    __builtin_amdgcn_s_wait_tensorcnt(0);
  }
#else
  for (int idx = tid; idx < 16 * 512; idx += 256)
    atile_f32[idx] = c_t[(size_t)m0 * C_SZ + idx];
#endif
  __syncthreads();

  // ---- Pack f32 tile -> bf16-pair tile (LDS -> LDS) ----------------------
  for (int idx = tid; idx < 16 * 256; idx += 256) {
    int row = idx >> 8, cp = idx & 255;
    atile[row * 260 + cp] =
        pack2(atile_f32[row * 512 + 2 * cp], atile_f32[row * 512 + 2 * cp + 1]);
  }
  __syncthreads();

  const int wave = tid >> 5;
  const int lane = tid & 31;
  const int half = lane >> 4;
  const int l16  = lane & 15;
  const int n0   = wave * 64;

  v8f zacc;
#pragma unroll
  for (int r = 0; r < 8; ++r) zacc[r] = 0.0f;
  v8f accP[4], accA[4];
#pragma unroll
  for (int nt = 0; nt < 4; ++nt) { accP[nt] = zacc; accA[nt] = zacc; }

  for (int ks = 0; ks < 16; ++ks) {          // K = 512 in steps of 32
    const int k0 = ks * 32;
    // A fragment, contiguous k order: lane half selects K+16.
    BFrag a;
#pragma unroll
    for (int i = 0; i < 8; ++i) {
      int kb = k0 + 2 * i + 16 * half;
      a.u[i] = atile[l16 * 260 + (kb >> 1)];
    }
    // Load ALL B fragments for this K-step first (one clause / one wait),
    // then issue the 8 WMMAs back-to-back.
    BFrag bP[4], bA[4];
#pragma unroll
    for (int nt = 0; nt < 4; ++nt) {
      const int n = n0 + nt * 16 + l16;
#pragma unroll
      for (int i = 0; i < 8; ++i) {
        int kb = k0 + 2 * i + 16 * half;
        bP[nt].u[i] = *(const unsigned*)(wp_bf + (size_t)n * C_SZ + kb);
        bA[nt].u[i] = *(const unsigned*)(wa_t  + (size_t)n * C_SZ + kb);
      }
    }
#pragma unroll
    for (int nt = 0; nt < 4; ++nt) {
      accP[nt] = __builtin_amdgcn_wmma_f32_16x16x32_bf16(
          false, a.v, false, bP[nt].v, (short)0, accP[nt], false, false);
      accA[nt] = __builtin_amdgcn_wmma_f32_16x16x32_bf16(
          false, a.v, false, bA[nt].v, (short)0, accA[nt], false, false);
    }
  }

  // Epilogue A: store u = c_t @ W_a (f32). C/D layout: m = r + 8*half, n = l16.
#pragma unroll
  for (int nt = 0; nt < 4; ++nt) {
    const int n = n0 + nt * 16 + l16;
#pragma unroll
    for (int r = 0; r < 8; ++r) {
      u[(size_t)(m0 + r + 8 * half) * C_SZ + n] = accA[nt][r];
    }
  }

  // Epilogue P: p_t = S * sigmoid( V_p . tanh(h) ), reduced via LDS atomics.
  float part[8];
#pragma unroll
  for (int r = 0; r < 8; ++r) part[r] = 0.0f;
#pragma unroll
  for (int nt = 0; nt < 4; ++nt) {
    const float vp = V_p[n0 + nt * 16 + l16];
#pragma unroll
    for (int r = 0; r < 8; ++r) part[r] += tanhf(accP[nt][r]) * vp;
  }
#pragma unroll
  for (int r = 0; r < 8; ++r) atomicAdd(&red[r + 8 * half], part[r]);
  __syncthreads();
  if (tid < 16)
    p_t[m0 + tid] = (float)S_SZ / (1.0f + __expf(-red[tid]));
}

// ---------------------------------------------------------------------------
// Kernel 3: gathered local attention. One wave32 per (b,t) row.
// Gathers the 7 q-rows once into registers (re-used for dots and output),
// wave reductions via shfl_xor (ds_bpermute path on gfx1250 wave32).
// ---------------------------------------------------------------------------
__global__ __launch_bounds__(256) void attn_kernel(
    const float* __restrict__ q, const float* __restrict__ u,
    const float* __restrict__ p_t, float* __restrict__ out) {
  const int wave = threadIdx.x >> 5;
  const int lane = threadIdx.x & 31;
  const int m = blockIdx.x * 8 + wave;        // 2048 blocks * 8 waves = 16384
  const int b = m >> 10;                      // T = 1024
  const float* qb = q + (size_t)b * C_SZ * S_SZ;

  const float p = p_t[m];
  const int ip = (int)p;                      // trunc, p in (0, 4096)

  int  sw[WIN];
  bool msk[WIN];
#pragma unroll
  for (int w = 0; w < WIN; ++w) {
    int s = ip + w - 2;                       // offsets -2..+4 (D=3)
    s = min(max(s, 0), S_SZ + 1) % (S_SZ + 1);
    sw[w] = s;
    msk[w] = (s == 0);                        // NaN-pad position
  }

  float qreg[16][WIN];
  float acc[WIN];
#pragma unroll
  for (int w = 0; w < WIN; ++w) acc[w] = 0.0f;

#pragma unroll
  for (int j = 0; j < 16; ++j) {
    const int ch = lane + 32 * j;
    const float uv = u[(size_t)m * C_SZ + ch];
#pragma unroll
    for (int w = 0; w < WIN; ++w) {
      float qv = msk[w] ? 0.0f : qb[(size_t)ch * S_SZ + (sw[w] - 1)];
      qreg[j][w] = qv;
      acc[w] += qv * uv;
    }
  }

  // full-wave sum of the 7 logits
#pragma unroll
  for (int w = 0; w < WIN; ++w) {
#pragma unroll
    for (int off = 16; off >= 1; off >>= 1)
      acc[w] += __shfl_xor(acc[w], off, 32);
  }

  // masked softmax * truncated Gaussian (computed redundantly per lane)
  float mx = -1e30f;
#pragma unroll
  for (int w = 0; w < WIN; ++w) if (!msk[w]) mx = fmaxf(mx, acc[w]);
  float e[WIN], den = 0.0f;
#pragma unroll
  for (int w = 0; w < WIN; ++w) {
    e[w] = msk[w] ? 0.0f : __expf(acc[w] - mx);
    den += e[w];
  }
  const float inv = 1.0f / den;
  float wt[WIN];
#pragma unroll
  for (int w = 0; w < WIN; ++w) {
    float sf = fmaxf((float)sw[w] - 1.0f, 0.0f);
    float d  = (sf - p) * (1.0f / 3.0f);      // D = 3
    wt[w] = e[w] * inv * __expf(-2.0f * d * d);
  }

#pragma unroll
  for (int j = 0; j < 16; ++j) {
    float o = 0.0f;
#pragma unroll
    for (int w = 0; w < WIN; ++w) o += wt[w] * qreg[j][w];
    out[(size_t)m * C_SZ + lane + 32 * j] = o;
  }
}

// ---------------------------------------------------------------------------
// Launch. Inputs (setup_inputs order): q, c_t, W_a, W_p, V_p. Output f32.
// Workspace: [p_t 64KB][wp_bf 512KB][wa_t 512KB][u 32MB]
// ---------------------------------------------------------------------------
extern "C" void kernel_launch(void* const* d_in, const int* in_sizes, int n_in,
                              void* d_out, int out_size, void* d_ws, size_t ws_size,
                              hipStream_t stream) {
  const float* q   = (const float*)d_in[0];
  const float* c_t = (const float*)d_in[1];
  const float* W_a = (const float*)d_in[2];
  const float* W_p = (const float*)d_in[3];
  const float* V_p = (const float*)d_in[4];
  float* out = (float*)d_out;

  char* ws = (char*)d_ws;
  float*          p_t   = (float*)ws;                              // 16384 f32
  unsigned short* wp_bf = (unsigned short*)(ws + (64 << 10));      // 512 KB
  unsigned short* wa_t  = (unsigned short*)(ws + (64 << 10) + (512 << 10));
  float*          u     = (float*)(ws + (64 << 10) + (1024 << 10));// 32 MB

  prep_kernel<<<(C_SZ * C_SZ) / 256, 256, 0, stream>>>(W_p, W_a, wp_bf, wa_t);
  fused_gemm_kernel<<<M_SZ / 16, 256, 0, stream>>>(c_t, V_p, wp_bf, wa_t, u, p_t);
  attn_kernel<<<M_SZ / 8, 256, 0, stream>>>(q, u, p_t, out);
}
